// CrossAttention_17738214932658
// MI455X (gfx1250) — matrix operand order repaired and checked by reading, not compiled, not run on hardware
//
#include <hip/hip_runtime.h>

#define B_    4
#define QLEN  2048
#define KLEN  2048
#define D_    128
#define H_    4
#define DH    32
#define EPSLN 1e-5f
#define QSCALE 0.17677669529663687f   /* 32^-0.5 */

#define USE_ASYNC_LDS 1

typedef __attribute__((ext_vector_type(16))) _Float16 v16h;
typedef __attribute__((ext_vector_type(8)))  _Float16 v8h;
typedef __attribute__((ext_vector_type(8)))  float    v8f;

__device__ __forceinline__ float sigm(float x) { return 1.f / (1.f + __expf(-x)); }

__device__ __forceinline__ v8f wmma_f16(v16h a, v16h b, v8f c) {
  return __builtin_amdgcn_wmma_f32_16x16x32_f16(false, a, false, b, (short)0, c, false, false);
}

__device__ __forceinline__ v16h join16(v8h lo, v8h hi) {
  v16h r;
#pragma unroll
  for (int j = 0; j < 8; ++j) { r[j] = lo[j]; r[8 + j] = hi[j]; }
  return r;
}

// A fragment (16x32 f16) from row-major f16 LDS tile (row stride `ld` halves,
// ld*2 % 16 == 0 so both 8-half chunks are 16B aligned -> ds_load_b128).
__device__ __forceinline__ v16h ldsA(const _Float16* base, int ld, int lane, int kbase) {
  int g = lane >> 4, r = lane & 15;
  const _Float16* p = base + r * ld + kbase + 16 * g;
  return join16(*(const v8h*)p, *(const v8h*)(p + 8));
}

// B fragment (32x16) from f32 weight W[d][n] ([in,out]); K-chunk kbase, N tile n0.
__device__ __forceinline__ v16h wB(const float* W, int lane, int kbase, int n0) {
  int g = lane >> 4, n = lane & 15;
  const float* p = W + (size_t)(kbase + 16 * g) * D_ + n0 + n;
  v16h b;
#pragma unroll
  for (int j = 0; j < 16; ++j) b[j] = (_Float16)p[(size_t)j * D_];
  return b;
}

#if USE_ASYNC_LDS
// GLOBAL_LOAD_ASYNC_TO_LDS_B128: per-lane 16B DMA global->LDS, tracked by ASYNCcnt.
__device__ __forceinline__ void async_ld16(const void* g, void* l) {
  unsigned loff = (unsigned)(unsigned long long)(uintptr_t)l;
  unsigned long long ga = (unsigned long long)(uintptr_t)g;
  asm volatile("global_load_async_to_lds_b128 %0, %1, off" :: "v"(loff), "v"(ga) : "memory");
}
__device__ __forceinline__ void wait_async0() { asm volatile("s_wait_asynccnt 0x0" ::: "memory"); }
__device__ __forceinline__ void wait_async2() { asm volatile("s_wait_asynccnt 0x2" ::: "memory"); }
#else
__device__ __forceinline__ void wait_async0() {}
__device__ __forceinline__ void wait_async2() {}
#endif

// ---------------- Kernel 1/2: fused AdaLN + projections ----------------
// MODE 0 (q side): out1_h = (xq@Wq+bq)*QSCALE, out2_f = xq@Wg logits, out3_f = sigmoid(cond@Wz+bz)
// MODE 1 (k side): out1_h = xk@Wk (row-major), out2_h = xk@Wv stored TRANSPOSED as
//                  vhT[((b*H+h)*DH+dh)*KLEN + k]  (single v8h store per lane)
template <int MODE>
__global__ __launch_bounds__(256) void prep_kernel(
    const float* __restrict__ x, const float* __restrict__ cond,
    const float* __restrict__ ln_g, const float* __restrict__ scale_W,
    const float* __restrict__ scale_b, const float* __restrict__ bias_W,
    const float* __restrict__ W1, const float* __restrict__ b1,
    const float* __restrict__ W2, const float* __restrict__ W3,
    const float* __restrict__ b3, _Float16* __restrict__ out1_h,
    _Float16* __restrict__ out2_h, float* __restrict__ out2_f,
    float* __restrict__ out3_f) {
  __shared__ float xs[16][132];
  __shared__ float cs[16][132];
  __shared__ __align__(16) _Float16 xn[16][136];
  __shared__ __align__(16) _Float16 cn[16][136];
  __shared__ __align__(16) _Float16 cr[16][136];
  __shared__ __align__(16) _Float16 xa[16][136];

  const int t = threadIdx.x;
  const int row0 = blockIdx.x * 16;
  {
    int r = t >> 4, c0 = (t & 15) * 8;
    const float* xp = x + (size_t)(row0 + r) * D_ + c0;
    const float* cp = cond + (size_t)(row0 + r) * D_ + c0;
#pragma unroll
    for (int j = 0; j < 8; ++j) { xs[r][c0 + j] = xp[j]; cs[r][c0 + j] = cp[j]; }
  }
  __syncthreads();
  {  // LayerNorm: 16 threads/row, shfl_xor tree within the 16-lane half-wave
    int r = t >> 4, c0 = (t & 15) * 8;
    float sx = 0, sx2 = 0, sc = 0, sc2 = 0;
#pragma unroll
    for (int j = 0; j < 8; ++j) {
      float v = xs[r][c0 + j]; sx += v; sx2 += v * v;
      float w = cs[r][c0 + j]; sc += w; sc2 += w * w;
    }
#pragma unroll
    for (int m = 1; m < 16; m <<= 1) {
      sx += __shfl_xor(sx, m, 32);  sx2 += __shfl_xor(sx2, m, 32);
      sc += __shfl_xor(sc, m, 32);  sc2 += __shfl_xor(sc2, m, 32);
    }
    float mx = sx / D_, vx = sx2 / D_ - mx * mx;
    float mc = sc / D_, vc = sc2 / D_ - mc * mc;
    float rx = rsqrtf(vx + EPSLN), rc = rsqrtf(vc + EPSLN);
#pragma unroll
    for (int j = 0; j < 8; ++j) {
      int c = c0 + j;
      xn[r][c] = (_Float16)((xs[r][c] - mx) * rx);
      cn[r][c] = (_Float16)((cs[r][c] - mc) * rc * ln_g[c]);
      cr[r][c] = (_Float16)cs[r][c];
    }
  }
  __syncthreads();

  const int wave = t >> 5, lane = t & 31;
  const int n0 = wave * 16;
  const int g = lane >> 4, n = lane & 15;

  v8f accS = {}, accB = {};
#pragma unroll
  for (int ks = 0; ks < 4; ++ks) {
    v16h a = ldsA(&cn[0][0], 136, lane, ks * 32);
    accS = wmma_f16(a, wB(scale_W, lane, ks * 32, n0), accS);
    accB = wmma_f16(a, wB(bias_W, lane, ks * 32, n0), accB);
  }
#pragma unroll
  for (int i = 0; i < 8; ++i) {
    int m = i + 8 * g;
    float xnv = (float)xn[m][n0 + n];
    xa[m][n0 + n] = (_Float16)(sigm(accS[i] + scale_b[n0 + n]) * xnv + accB[i]);
  }
  __syncthreads();

  v8f acc1 = {}, acc2 = {};
#pragma unroll
  for (int ks = 0; ks < 4; ++ks) {
    v16h a = ldsA(&xa[0][0], 136, lane, ks * 32);
    acc1 = wmma_f16(a, wB(W1, lane, ks * 32, n0), acc1);
    acc2 = wmma_f16(a, wB(W2, lane, ks * 32, n0), acc2);
  }
  if (MODE == 0) {
    v8f acc3 = {};
#pragma unroll
    for (int ks = 0; ks < 4; ++ks) {
      v16h a = ldsA(&cr[0][0], 136, lane, ks * 32);
      acc3 = wmma_f16(a, wB(W3, lane, ks * 32, n0), acc3);
    }
#pragma unroll
    for (int i = 0; i < 8; ++i) {
      int m = i + 8 * g; size_t row = row0 + m; int c = n0 + n;
      out1_h[row * D_ + c] = (_Float16)((acc1[i] + b1[c]) * QSCALE);
      out2_f[row * D_ + c] = acc2[i];
      out3_f[row * D_ + c] = sigm(acc3[i] + b3[c]);
    }
  } else {
#pragma unroll
    for (int i = 0; i < 8; ++i) {
      int m = i + 8 * g; size_t row = row0 + m; int c = n0 + n;
      out1_h[row * D_ + c] = (_Float16)acc1[i];
    }
    // transposed V store: this lane's 8 rows are 8 consecutive k's -> one 16B store
    {
      int c = n0 + n;
      int b = row0 / KLEN;
      int kbase = row0 - b * KLEN + 8 * g;
      v8h pk;
#pragma unroll
      for (int i = 0; i < 8; ++i) pk[i] = (_Float16)acc2[i];
      *(v8h*)&out2_h[((size_t)(b * H_ + (c >> 5)) * DH + (c & 31)) * KLEN + kbase] = pk;
    }
  }
}

// ---------------- Kernel 3: flash attention with pair bias ----------------
// grid.x = B*H, grid.y = Q/128; 8 waves/block, one 16-row q-tile per wave.
// K-chunk = 64; K and transposed-V tiles double-buffered via async DMA to LDS.
__global__ __launch_bounds__(256) void attn_kernel(
    const _Float16* __restrict__ qh, const _Float16* __restrict__ kh,
    const _Float16* __restrict__ vhT, const float* __restrict__ pair,
    const unsigned char* __restrict__ mask_q,
    const unsigned char* __restrict__ mask_k, float* __restrict__ attn) {
  __shared__ __align__(16) _Float16 kt[2][64][40];    // K rows x DH
  __shared__ __align__(16) _Float16 vtT[2][32][72];   // V transposed: [dh][k]
  __shared__ __align__(16) _Float16 pt[8][16][72];    // per-wave P (16x64)

  const int bh = blockIdx.x;
  const int b = bh / H_, h = bh % H_;
  const int t = threadIdx.x;
  const int wave = t >> 5, lane = t & 31;
  const int g = lane >> 4, n = lane & 15;
  const int q0 = blockIdx.y * 128 + wave * 16;

  const int tr = t >> 2, tc = (t & 3) * 8;   // K-tile chunk: row 0..63, 8 halves
  const int vdh = t >> 3, vch = (t & 7) * 8; // V-tile chunk: dh 0..31, 8 halves

  // Q A-fragment (qh already holds *QSCALE)
  v16h qa;
  {
    int r = lane & 15;
    const _Float16* p = qh + ((size_t)(b * QLEN) + q0 + r) * D_ + h * DH + 16 * g;
    qa = join16(*(const v8h*)p, *(const v8h*)(p + 8));
  }
  float nmq[8];
#pragma unroll
  for (int i = 0; i < 8; ++i)
    nmq[i] = mask_q[b * QLEN + q0 + i + 8 * g] ? 0.f : 1.f;

  float run_m[8], lsum[8];
  v8f acc0 = {}, acc1 = {};
#pragma unroll
  for (int i = 0; i < 8; ++i) { run_m[i] = -1e30f; lsum[i] = 0.f; }

  auto issue_tile = [&](int it) {
    int buf = it & 1, k0 = it * 64;
    const _Float16* kp = kh + ((size_t)(b * KLEN) + k0 + tr) * D_ + h * DH + tc;
    const _Float16* vp = vhT + ((size_t)(b * H_ + h) * DH + vdh) * KLEN + k0 + vch;
#if USE_ASYNC_LDS
    async_ld16(kp, &kt[buf][tr][tc]);
    async_ld16(vp, &vtT[buf][vdh][vch]);
#else
    *(uint4*)&kt[buf][tr][tc] = *(const uint4*)kp;
    *(uint4*)&vtT[buf][vdh][vch] = *(const uint4*)vp;
#endif
  };

  const int NT = KLEN / 64;
  issue_tile(0);
  for (int it = 0; it < NT; ++it) {
    const int buf = it & 1, k0 = it * 64;
    if (it + 1 < NT) { issue_tile(it + 1); wait_async2(); }
    else            { wait_async0(); }
    __syncthreads();  // tile `it` visible to all waves

    // ---- logits: four 16x16 sub-tiles; C seeded with pair_logits + mask bias ----
    v8f s[4];
#pragma unroll
    for (int kk = 0; kk < 4; ++kk) {
      int kidx = k0 + kk * 16 + n;
      float nmk = mask_k[b * KLEN + kidx] ? 0.f : 1.f;
      v8f c;
#pragma unroll
      for (int i = 0; i < 8; ++i) {
        int m = i + 8 * g;
        c[i] = pair[(((size_t)(b * H_ + h) * QLEN + q0 + m) * KLEN) + kidx] +
               1e9f * nmq[i] * nmk;
      }
      const _Float16* kp = &kt[buf][kk * 16 + n][16 * g];
      v16h bf = join16(*(const v8h*)kp, *(const v8h*)(kp + 8));
      s[kk] = wmma_f16(qa, bf, c);
    }

    // ---- online softmax: only the MAX needs a cross-lane butterfly; the row-sum
    // is kept as a per-lane partial (corr is row-uniform) and reduced after the loop.
    float corr[8];
#pragma unroll
    for (int i = 0; i < 8; ++i) {
      float v = fmaxf(fmaxf(s[0][i], s[1][i]), fmaxf(s[2][i], s[3][i]));
#pragma unroll
      for (int m2 = 1; m2 < 16; m2 <<= 1) v = fmaxf(v, __shfl_xor(v, m2, 32));
      float nm = fmaxf(run_m[i], v);
      corr[i] = __expf(run_m[i] - nm);
      run_m[i] = nm;
    }
#pragma unroll
    for (int i = 0; i < 8; ++i) {
      int m = i + 8 * g;
      float ps = 0.f;
#pragma unroll
      for (int kk = 0; kk < 4; ++kk) {
        float p = __expf(s[kk][i] - run_m[i]);
        ps += p;
        pt[wave][m][kk * 16 + n] = (_Float16)p;
      }
      lsum[i] = lsum[i] * corr[i] + ps;
      acc0[i] *= corr[i];
      acc1[i] *= corr[i];
    }

    // ---- P(16x64) @ V(64x32): 2 K-steps x 2 N-tiles ----
#pragma unroll
    for (int kc = 0; kc < 2; ++kc) {
      v16h pa = ldsA(&pt[wave][0][0], 72, lane, kc * 32);
      const _Float16* v0 = &vtT[buf][n][kc * 32 + 16 * g];
      const _Float16* v1 = &vtT[buf][16 + n][kc * 32 + 16 * g];
      acc0 = wmma_f16(pa, join16(*(const v8h*)v0, *(const v8h*)(v0 + 8)), acc0);
      acc1 = wmma_f16(pa, join16(*(const v8h*)v1, *(const v8h*)(v1 + 8)), acc1);
    }
    __syncthreads();  // all waves done with buf before tile it+2 overwrites it
  }

  // deferred l reduction (once) + store
#pragma unroll
  for (int i = 0; i < 8; ++i) {
    int m = i + 8 * g;
    float l = lsum[i];
#pragma unroll
    for (int m2 = 1; m2 < 16; m2 <<= 1) l += __shfl_xor(l, m2, 32);
    float inv = 1.f / l;
    attn[((size_t)(b * QLEN) + q0 + m) * D_ + h * DH + n] = acc0[i] * inv;
    attn[((size_t)(b * QLEN) + q0 + m) * D_ + h * DH + 16 + n] = acc1[i] * inv;
  }
}

// ---------------- Kernel 4: gated output projection ----------------
__global__ __launch_bounds__(256) void epilogue_kernel(
    const float* __restrict__ attn, const float* __restrict__ gl,
    const float* __restrict__ zsig, const float* __restrict__ Wt,
    float* __restrict__ y) {
  __shared__ __align__(16) _Float16 ob[16][136];
  const int t = threadIdx.x;
  const int row0 = blockIdx.x * 16;
  {
    int r = t >> 4, c0 = (t & 15) * 8;
    const float* ap = attn + (size_t)(row0 + r) * D_ + c0;
    const float* gp = gl + (size_t)(row0 + r) * D_ + c0;
#pragma unroll
    for (int j = 0; j < 8; ++j) ob[r][c0 + j] = (_Float16)(ap[j] * sigm(gp[j]));
  }
  __syncthreads();
  const int wave = t >> 5, lane = t & 31;
  const int n0 = wave * 16, g = lane >> 4, n = lane & 15;
  v8f acc = {};
#pragma unroll
  for (int ks = 0; ks < 4; ++ks)
    acc = wmma_f16(ldsA(&ob[0][0], 136, lane, ks * 32), wB(Wt, lane, ks * 32, n0), acc);
#pragma unroll
  for (int i = 0; i < 8; ++i) {
    size_t row = row0 + i + 8 * g; int c = n0 + n;
    y[row * D_ + c] = acc[i] * zsig[row * D_ + c];
  }
}

extern "C" void kernel_launch(void* const* d_in, const int* in_sizes, int n_in,
                              void* d_out, int out_size, void* d_ws, size_t ws_size,
                              hipStream_t stream) {
  (void)in_sizes; (void)n_in; (void)out_size; (void)ws_size;
  const float* x_q = (const float*)d_in[0];
  const float* x_k = (const float*)d_in[1];
  const unsigned char* mask_q = (const unsigned char*)d_in[2];
  const unsigned char* mask_k = (const unsigned char*)d_in[3];
  const float* pair = (const float*)d_in[4];
  const float* cond_q = (const float*)d_in[5];
  const float* cond_k = (const float*)d_in[6];
  const float* q_ln_g = (const float*)d_in[7];
  const float* q_scale_W = (const float*)d_in[8];
  const float* q_scale_b = (const float*)d_in[9];
  const float* q_bias_W = (const float*)d_in[10];
  const float* k_ln_g = (const float*)d_in[11];
  const float* k_scale_W = (const float*)d_in[12];
  const float* k_scale_b = (const float*)d_in[13];
  const float* k_bias_W = (const float*)d_in[14];
  const float* Wq = (const float*)d_in[15];
  const float* bq = (const float*)d_in[16];
  const float* Wk = (const float*)d_in[17];
  const float* Wv = (const float*)d_in[18];
  const float* Wg = (const float*)d_in[19];
  const float* Wt = (const float*)d_in[20];
  const float* Wz = (const float*)d_in[21];
  const float* bz = (const float*)d_in[22];

  char* ws = (char*)d_ws;
  size_t off = 0;
  _Float16* qh  = (_Float16*)(ws + off); off += (size_t)B_ * QLEN * D_ * 2;
  _Float16* kh  = (_Float16*)(ws + off); off += (size_t)B_ * KLEN * D_ * 2;
  _Float16* vhT = (_Float16*)(ws + off); off += (size_t)B_ * KLEN * D_ * 2;
  float* gl   = (float*)(ws + off); off += (size_t)B_ * QLEN * D_ * 4;
  float* zsig = (float*)(ws + off); off += (size_t)B_ * QLEN * D_ * 4;
  float* attn = (float*)(ws + off); off += (size_t)B_ * QLEN * D_ * 4;

  dim3 blk(256);
  prep_kernel<0><<<B_ * QLEN / 16, blk, 0, stream>>>(
      x_q, cond_q, q_ln_g, q_scale_W, q_scale_b, q_bias_W, Wq, bq, Wg, Wz, bz,
      qh, nullptr, gl, zsig);
  prep_kernel<1><<<B_ * KLEN / 16, blk, 0, stream>>>(
      x_k, cond_k, k_ln_g, k_scale_W, k_scale_b, k_bias_W, Wk, nullptr, Wv,
      nullptr, nullptr, kh, vhT, nullptr, nullptr);
  dim3 agrid(B_ * H_, QLEN / 128);
  attn_kernel<<<agrid, blk, 0, stream>>>(qh, kh, vhT, pair, mask_q, mask_k, attn);
  epilogue_kernel<<<B_ * QLEN / 16, blk, 0, stream>>>(attn, gl, zsig, Wt,
                                                      (float*)d_out);
}
